// SCF_GRU_58420145160918
// MI455X (gfx1250) — matrix operand order repaired and checked by reading, not compiled, not run on hardware
//
#include <hip/hip_runtime.h>
#include <hip/hip_bf16.h>
#include <math.h>

#define B_    64
#define A_    64
#define T_    40
#define HID_  48
#define N_    (B_*A_)        // 4096
#define M_    64
#define MHID_ (M_*HID_)      // 3072
#define NT_   (MHID_/16)     // 192 column tiles
#define IMGH_ 80
#define R0_   0.5f
#define R1_   32.0f
#define RSTEP_  ((R1_-R0_)/8.0f)           // 3.9375
#define THSTEP_ (6.2831853071795864769f/8.0f)
#define TWO_PI_ 6.2831853071795864769f

typedef __attribute__((ext_vector_type(16))) __bf16 v16bf;
typedef __attribute__((ext_vector_type(8)))  float  v8f;

// ---------------------------------------------------------------- utilities
__global__ void zero_f32_kernel(float* __restrict__ p, int n) {
    int i = blockIdx.x * blockDim.x + threadIdx.x;
    if (i < n) p[i] = 0.0f;
}

// Repack fc_w (48 x 3072, row-major, col=(m,d)) directly into WMMA
// B-fragment order of this file (contiguous k, as for A) for v_wmma_f32_16x16x32_bf16:
//   BmatF[((tileN*2 + ks)*32 + lane)*16 + j] = B[k][n]
//   with k = ks*32 + 16*(lane/16) + j   (zero-padded K 48->64)
//        n = tileN*16 + lane%16
//   B[k=d][n=(m,d')] = fc_w[d'][m*48+d]
// Each lane's 16 bf16 are contiguous (32B) -> 2x global_load_b128 per frag.
__global__ void repack_fc_kernel(const float* __restrict__ fc_w,
                                 __bf16* __restrict__ BmatF) {
    int idx = blockIdx.x * blockDim.x + threadIdx.x;   // 192*2*32*16 = 196608
    if (idx >= NT_ * 2 * 32 * 16) return;
    int j    = idx & 15;
    int lane = (idx >> 4) & 31;
    int ks   = (idx >> 9) & 1;
    int tn   = idx >> 10;
    int k = ks * 32 + ((lane >> 4) << 4) + j;
    int n = tn * 16 + (lane & 15);
    float v = 0.0f;
    if (k < HID_) {
        int m  = n / HID_;
        int dp = n % HID_;
        v = fc_w[dp * MHID_ + m * HID_ + k];
    }
    BmatF[idx] = (__bf16)v;
}

// ------------------------------------------------------- pair bin precompute
// One block of 64 threads per (timestep, b, j). Packs mask|bin|count into u16.
__global__ void bins_kernel(const float* __restrict__ Y_path,
                            unsigned short* __restrict__ binpack, int t0) {
    int bid  = blockIdx.x;
    int pair = bid & (N_ - 1);
    int ts   = t0 + (bid >> 12);
    int b    = pair >> 6;
    int j    = pair & 63;
    int tt   = threadIdx.x;          // neighbor index 0..63

    __shared__ int cnt[64];
    cnt[tt] = 0;
    __syncthreads();

    float ljx = Y_path[((size_t)ts * N_ + pair) * 2 + 0];
    float ljy = Y_path[((size_t)ts * N_ + pair) * 2 + 1];
    float ltx = Y_path[((size_t)ts * N_ + b * 64 + tt) * 2 + 0];
    float lty = Y_path[((size_t)ts * N_ + b * 64 + tt) * 2 + 1];
    float cx = ltx - ljx, cy = lty - ljy;
    float dist = sqrtf(cx * cx + cy * cy);
    float safe = fmaxf(dist, 1e-12f);
    float cost = fminf(fmaxf(cx / safe, -1.0f), 1.0f);
    float ac = acosf(cost);
    float theta = (cy < 0.0f) ? (TWO_PI_ - ac) : ac;
    bool mask = (dist >= R0_) && (dist <= R1_) && (tt != j);
    int u = min(max((int)floorf((dist - R0_) / RSTEP_), 0), 7);
    int v = min(max((int)floorf(theta / THSTEP_), 0), 7);
    int bin = u * 8 + v;
    if (mask) atomicAdd(&cnt[bin], 1);
    __syncthreads();
    int c = min(cnt[bin], 63);
    binpack[(size_t)bid * 64 + tt] =
        (unsigned short)((mask ? 1 : 0) | (bin << 1) | (c << 7));
}

// ------------------------------------------------------ G GEMM (WMMA, bf16)
// G[tt, (m,d')] = sum_d state[i0+tt, d] * fc_w[d', m*48+d]
// M=64, N=3072, K=48 (padded 64). One wave per 16x16 tile: 768 waves.
// B fragments are pre-packed in register order -> straight 32B vector loads.
__global__ void g_gemm_kernel(const float* __restrict__ state, int i0,
                              const __bf16* __restrict__ BmatF,
                              float* __restrict__ G) {
    int wid  = (blockIdx.x * blockDim.x + threadIdx.x) >> 5;  // 0..767
    int lane = threadIdx.x & 31;
    int tileN = wid % NT_;                    // 0..191
    int tileM = wid / NT_;                    // 0..3
    int m0 = tileM * 16, n0 = tileN * 16;
    int half = lane >> 4, l16 = lane & 15;
    int arow = i0 + m0 + l16;                 // A-matrix row (state row)

    v8f acc = {};
    #pragma unroll
    for (int ks = 0; ks < 2; ++ks) {
        v16bf a;
        #pragma unroll
        for (int j = 0; j < 16; ++j) {
            // A element j of lane -> K = 32*ks + 16*half + j
            int ka = ks * 32 + (half << 4) + j;
            float av = (ka < HID_) ? state[(size_t)arow * HID_ + ka] : 0.0f;
            a[j] = (__bf16)av;
        }
        // pre-packed fragment: 16 contiguous bf16 per lane (32B)
        v16bf b = *(const v16bf*)(BmatF +
                   (((size_t)(tileN * 2 + ks) * 32 + lane) << 4));
        acc = __builtin_amdgcn_wmma_f32_16x16x32_bf16(
            false, a, false, b, (short)0, acc, false, false);
    }
    #pragma unroll
    for (int r = 0; r < 8; ++r) {
        int m = m0 + r + half * 8;            // D VGPR r -> M = r + 8*half
        G[(size_t)m * MHID_ + n0 + l16] = acc[r];
    }
}

// --------------------------------------------- fused pooling + GRU step
__device__ __forceinline__ float block_sum128(float v, float* red, int tid) {
    red[tid] = v;
    __syncthreads();
    #pragma unroll
    for (int s = 64; s >= 1; s >>= 1) {
        if (tid < s) red[tid] += red[tid + s];
        __syncthreads();
    }
    float r = red[0];
    __syncthreads();
    return r;
}

__global__ void step_kernel(const unsigned short* __restrict__ binpack, // [N_][64]
                            const float* __restrict__ G,                // [64][3072]
                            const float* __restrict__ state_old,        // [N_][48]
                            float* __restrict__ state_new,
                            float* __restrict__ out,                    // full d_out
                            const float* __restrict__ Y_path,
                            const float* __restrict__ Y_fv,
                            const float* __restrict__ feature_map,
                            const float* __restrict__ w_ir, const float* __restrict__ w_hr,
                            const float* __restrict__ b_ir, const float* __restrict__ b_hr,
                            const float* __restrict__ w_iz, const float* __restrict__ w_hz,
                            const float* __restrict__ b_iz, const float* __restrict__ b_hz,
                            const float* __restrict__ w_in, const float* __restrict__ w_hn,
                            const float* __restrict__ b_in, const float* __restrict__ b_hn,
                            const float* __restrict__ fc_b,
                            int t, int is_last) {
    __shared__ float sw[64];
    __shared__ int   sbin[64];
    __shared__ float sx[96];     // [feat 0..31 | fvel 32..47 | fsp 48..95]
    __shared__ float sh[48];
    __shared__ float red[128];

    int pair = blockIdx.x;
    int b    = pair >> 6;
    int tid  = threadIdx.x;      // 128 threads

    float locx = Y_path[((size_t)t * N_ + pair) * 2 + 0];
    float locy = Y_path[((size_t)t * N_ + pair) * 2 + 1];
    int ui = min(max(IMGH_ / 2 - (int)floorf(locy), 0), IMGH_ - 1);
    int vi = min(max((int)floorf(locx), 0), IMGH_ - 1);

    if (tid < 64) {
        unsigned short p = binpack[(size_t)pair * 64 + tid];
        int mask = p & 1;
        int bin  = (p >> 1) & 63;
        int cnt  = (p >> 7) & 63;
        sbin[tid] = bin;
        sw[tid]   = mask ? ((cnt > 1) ? (1.0f / (float)cnt) : 1.0f) : 0.0f;
    } else if (tid < 112) {
        sh[tid - 64] = state_old[(size_t)pair * HID_ + (tid - 64)];
    }
    __syncthreads();

    if (tid < 48) {
        float acc = fc_b[tid];
        for (int tt = 0; tt < 64; ++tt) {
            float ww = sw[tt];
            if (ww != 0.0f)
                acc = fmaf(ww, G[(size_t)tt * MHID_ + sbin[tt] * HID_ + tid], acc);
        }
        sx[48 + tid] = fmaxf(acc, 0.0f);                        // fsp (relu)
    } else if (tid < 80) {
        int c = tid - 48;
        sx[c] = feature_map[(((size_t)b * 32 + c) * IMGH_ + ui) * IMGH_ + vi];
    } else if (tid < 96) {
        int k = tid - 80;
        sx[32 + k] = Y_fv[((size_t)t * N_ + pair) * 16 + k];
    }
    __syncthreads();

    float ar = 0.f, az = 0.f, axn = 0.f, ahn = 0.f;
    if (tid < 96) {
        float xv = sx[tid];
        ar  = xv * w_ir[tid];
        az  = xv * w_iz[tid];
        axn = xv * w_in[tid];
    }
    if (tid < 48) {
        float hv = sh[tid];
        ar += hv * w_hr[tid];
        az += hv * w_hz[tid];
        ahn = hv * w_hn[tid];
    }
    float s_r  = block_sum128(ar,  red, tid);
    float s_z  = block_sum128(az,  red, tid);
    float s_xn = block_sum128(axn, red, tid);
    float s_hn = block_sum128(ahn, red, tid);

    float rt = 1.0f / (1.0f + expf(-(s_r + b_ir[0] + b_hr[0])));
    float zt = 1.0f / (1.0f + expf(-(s_z + b_iz[0] + b_hz[0])));
    float nt = tanhf(s_xn + b_in[0] + rt * (s_hn + b_hn[0]));

    if (tid < 48) {
        float ht = (1.0f - zt) * nt + zt * sh[tid];
        out[((size_t)t * N_ + pair) * HID_ + tid] = ht;
        state_new[(size_t)pair * HID_ + tid] = ht;
        if (is_last)
            out[(size_t)T_ * N_ * HID_ + (size_t)pair * HID_ + tid] = ht;
    }
}

// ---------------------------------------------------------------- launcher
extern "C" void kernel_launch(void* const* d_in, const int* in_sizes, int n_in,
                              void* d_out, int out_size, void* d_ws, size_t ws_size,
                              hipStream_t stream) {
    const float* Y_path      = (const float*)d_in[0];
    const float* Y_fv        = (const float*)d_in[1];
    const float* feature_map = (const float*)d_in[2];
    const float* w_ir = (const float*)d_in[3];
    const float* w_hr = (const float*)d_in[4];
    const float* b_ir = (const float*)d_in[5];
    const float* b_hr = (const float*)d_in[6];
    const float* w_iz = (const float*)d_in[7];
    const float* w_hz = (const float*)d_in[8];
    const float* b_iz = (const float*)d_in[9];
    const float* b_hz = (const float*)d_in[10];
    const float* w_in = (const float*)d_in[11];
    const float* w_hn = (const float*)d_in[12];
    const float* b_in = (const float*)d_in[13];
    const float* b_hn = (const float*)d_in[14];
    const float* fc_w = (const float*)d_in[15];
    const float* fc_b = (const float*)d_in[16];
    float* out = (float*)d_out;

    char* ws = (char*)d_ws;
    size_t off = 0;
    auto alloc = [&](size_t nbytes) -> void* {
        void* p = ws + off;
        off = (off + nbytes + 255) & ~(size_t)255;
        return p;
    };
    float*  state0 = (float*)alloc((size_t)N_ * HID_ * sizeof(float));
    float*  state1 = (float*)alloc((size_t)N_ * HID_ * sizeof(float));
    __bf16* BmatF  = (__bf16*)alloc((size_t)64 * MHID_ * sizeof(__bf16));
    float*  G      = (float*)alloc((size_t)64 * MHID_ * sizeof(float));

    size_t per_step_bins = (size_t)N_ * 64 * sizeof(unsigned short);
    size_t full_bins     = per_step_bins * T_;
    bool use_full = (off + full_bins) <= ws_size;
    unsigned short* bins =
        (unsigned short*)alloc(use_full ? full_bins : per_step_bins);

    // init h0 = 0
    zero_f32_kernel<<<(N_ * HID_ + 255) / 256, 256, 0, stream>>>(state0, N_ * HID_);
    // repack fc_w -> WMMA fragment-order bf16 (zero-padded K 48->64)
    repack_fc_kernel<<<(64 * MHID_ + 255) / 256, 256, 0, stream>>>(fc_w, BmatF);
    // pair bins are state-independent: precompute all T steps if ws allows
    if (use_full)
        bins_kernel<<<T_ * N_, 64, 0, stream>>>(Y_path, bins, 0);

    float* bufs[2] = {state0, state1};
    for (int t = 0; t < T_; ++t) {
        if (!use_full)
            bins_kernel<<<N_, 64, 0, stream>>>(Y_path, bins, t);
        // G = h_slice(rows t..t+63) @ fc  -- WMMA bf16
        g_gemm_kernel<<<96, 256, 0, stream>>>(bufs[t & 1], t, BmatF, G);
        const unsigned short* bt =
            use_full ? bins + (size_t)t * N_ * 64 : bins;
        step_kernel<<<N_, 128, 0, stream>>>(
            bt, G, bufs[t & 1], bufs[(t + 1) & 1], out,
            Y_path, Y_fv, feature_map,
            w_ir, w_hr, b_ir, b_hr, w_iz, w_hz, b_iz, b_hz,
            w_in, w_hn, b_in, b_hn, fc_b, t, (t == T_ - 1) ? 1 : 0);
    }
}
